// DecoderAttn_89464168776112
// MI455X (gfx1250) — matrix operand order repaired and checked by reading, not compiled, not run on hardware
//
#include <hip/hip_runtime.h>
#include <math.h>

// ---------------------------------------------------------------------------
// DecoderAttn for MI455X (gfx1250): bf16 WMMA GEMMs with async-LDS staged
// weights + memory-bound attention.  B=32, T=128, L=2048, D=1024, F=1024
// ---------------------------------------------------------------------------

typedef __attribute__((ext_vector_type(16))) __bf16        v16bf;
typedef __attribute__((ext_vector_type(8)))  float         v8f;
typedef __attribute__((ext_vector_type(4)))  unsigned int  v4u;
typedef __attribute__((ext_vector_type(4)))  float         v4f;

static __device__ __forceinline__ unsigned short f32_to_bf16(float f) {
  unsigned u = __float_as_uint(f);
  u += 0x7FFFu + ((u >> 16) & 1u);          // round-to-nearest-even
  return (unsigned short)(u >> 16);
}

// ---------------------------------------------------------------------------
// C = A (MxK, row-major bf16) * B^T (B row-major NxK bf16) + bias, fp32 out.
// Block = 256 thr (8 waves) covering 32(M) x 256(N).  Each wave owns a 16x64
// strip (4 WMMA accumulators, A fragment reused 4x).  The 256x32 bf16 B-tile
// is staged into LDS with GLOBAL_LOAD_ASYNC_TO_LDS_B128 (double-buffered,
// ASYNCcnt-throttled) and read back as fragments with DS_LOAD_B128 — both
// through inline asm so the LDS traffic is opaque to alias analysis (the
// buffer address never escapes as a pointer, only as LDS byte offsets).
// grid = (N/256, M/32).  Fragment k order (A and B alike): contiguous, k = hi*16 + s.
// ---------------------------------------------------------------------------
__global__ void k_gemm_bf16nt(const unsigned short* __restrict__ A,
                              const unsigned short* __restrict__ Bw,
                              const float* __restrict__ bias,
                              float* __restrict__ C,
                              int lda, int ldb, int ldc, int K)
{
  // 2 x (256 rows x 32 bf16) = 32 KB; single static LDS alloc -> offset 0.
  __shared__ unsigned short lbuf[2 * 256 * 32];
  // keep the LDS allocation alive (all accesses go through inline asm)
  asm volatile("" :: "v"(lbuf));

  const int tid  = threadIdx.x;
  const int lane = tid & 31;
  const int wave = tid >> 5;
  const int m0   = blockIdx.y * 32 + (wave >> 2) * 16;
  const int nblk = blockIdx.x * 256;
  const int nw   = (wave & 3) * 64;            // wave's n-offset within block
  const int r    = lane & 15;
  const int hi   = lane >> 4;

  // staging: thread copies 4 x 16B; rows i*64 + (tid>>2), 16B chunk tid&3
  const int srow  = tid >> 2;                  // 0..63
  const int spart = tid & 3;                   // 0..3

  const unsigned short* ap = A + (size_t)(m0 + r) * lda + hi * 16;

  v8f acc0 = {}, acc1 = {}, acc2 = {}, acc3 = {};

  auto stage = [&](int p, int k0) {
    const unsigned short* gbase = Bw + (size_t)nblk * ldb + k0 + spart * 8;
    const unsigned lbase = (unsigned)(p * 16384 + spart * 16);
#pragma unroll
    for (int i = 0; i < 4; ++i) {
      const unsigned short* gp = gbase + (size_t)(i * 64 + srow) * ldb;
      const unsigned lo = lbase + (unsigned)(i * 64 + srow) * 64;
      asm volatile("global_load_async_to_lds_b128 %0, %1, off"
                   :: "v"(lo), "v"(gp) : "memory");
    }
  };

  stage(0, 0);
  for (int k0 = 0; k0 < K; k0 += 32) {
    const int p  = (k0 >> 5) & 1;
    const int kn = (k0 + 32 < K) ? (k0 + 32) : k0;   // last iter: redundant restage
    stage(p ^ 1, kn);
    // <=4 outstanding: next tile's copies in flight, current tile complete
    asm volatile("s_wait_asynccnt 0x4" ::: "memory");
    __syncthreads();

    __builtin_prefetch((const void*)(ap + k0 + 256), 0, 0);
    union { v16bf v; v4u q[2]; } a;
    a.q[0] = *(const v4u*)(ap + k0);        // K = hi*16 + 0..7
    a.q[1] = *(const v4u*)(ap + k0 + 8);    // K = hi*16 + 8..15

    // B fragments: lane holds column (nw + s*16 + r), 16 contiguous K at
    // hi*16.  Row pitch = 64B, s-tile pitch = 16*64B = 1024B.
    const unsigned baddr = (unsigned)(p * 16384 + (nw + r) * 64 + hi * 32);
    v4u b00, b01, b10, b11, b20, b21, b30, b31;
    asm volatile(
        "ds_load_b128 %0, %8 offset:0\n\t"
        "ds_load_b128 %1, %8 offset:16\n\t"
        "ds_load_b128 %2, %8 offset:1024\n\t"
        "ds_load_b128 %3, %8 offset:1040\n\t"
        "ds_load_b128 %4, %8 offset:2048\n\t"
        "ds_load_b128 %5, %8 offset:2064\n\t"
        "ds_load_b128 %6, %8 offset:3072\n\t"
        "ds_load_b128 %7, %8 offset:3088\n\t"
        "s_wait_dscnt 0x0"
        : "=&v"(b00), "=&v"(b01), "=&v"(b10), "=&v"(b11),
          "=&v"(b20), "=&v"(b21), "=&v"(b30), "=&v"(b31)
        : "v"(baddr)
        : "memory");

    union { v16bf v; v4u q[2]; } b;
    b.q[0] = b00; b.q[1] = b01;
    acc0 = __builtin_amdgcn_wmma_f32_16x16x32_bf16(false, a.v, false, b.v,
                                                   (short)0, acc0, false, false);
    b.q[0] = b10; b.q[1] = b11;
    acc1 = __builtin_amdgcn_wmma_f32_16x16x32_bf16(false, a.v, false, b.v,
                                                   (short)0, acc1, false, false);
    b.q[0] = b20; b.q[1] = b21;
    acc2 = __builtin_amdgcn_wmma_f32_16x16x32_bf16(false, a.v, false, b.v,
                                                   (short)0, acc2, false, false);
    b.q[0] = b30; b.q[1] = b31;
    acc3 = __builtin_amdgcn_wmma_f32_16x16x32_bf16(false, a.v, false, b.v,
                                                   (short)0, acc3, false, false);

    __syncthreads();   // reads done before this buffer is restaged
  }

#pragma unroll
  for (int s = 0; s < 4; ++s) {
    const int n  = nblk + nw + s * 16 + r;
    const float bn = bias[n];
    const v8f& acc = (s == 0) ? acc0 : (s == 1) ? acc1 : (s == 2) ? acc2 : acc3;
#pragma unroll
    for (int j = 0; j < 8; ++j) {
      const int m = m0 + hi * 8 + j;
      C[(size_t)m * ldc + n] = acc[j] + bn;
    }
  }
}

// ---------------------------------------------------------------------------
// One RNN step: h_new = tanh(xW[:,t,:] + h_prev @ W_hh^T)  (M=32,N=1024,K=1024)
// Serial latency path: direct loads, no barriers.  grid=(16,1), block=256.
// ---------------------------------------------------------------------------
__global__ void k_step(const unsigned short* __restrict__ hbf_in,
                       const unsigned short* __restrict__ Whh,
                       const float* __restrict__ xW, int t,
                       float* __restrict__ hf_out,
                       unsigned short* __restrict__ hbf_out)
{
  const int lane = threadIdx.x & 31;
  const int wave = threadIdx.x >> 5;
  const int m0 = (wave >> 2) * 16;           // M = 32 -> two m tiles
  const int n0 = blockIdx.x * 64 + (wave & 3) * 16;
  const int r  = lane & 15;
  const int hi = lane >> 4;

  const unsigned short* ap = hbf_in + (size_t)(m0 + r) * 1024 + hi * 16;
  const unsigned short* bp = Whh    + (size_t)(n0 + r) * 1024 + hi * 16;

  v8f c = {};
  for (int k0 = 0; k0 < 1024; k0 += 32) {
    union { v16bf v; v4u q[2]; } a, b;
    a.q[0] = *(const v4u*)(ap + k0);
    a.q[1] = *(const v4u*)(ap + k0 + 8);
    b.q[0] = *(const v4u*)(bp + k0);
    b.q[1] = *(const v4u*)(bp + k0 + 8);
    c = __builtin_amdgcn_wmma_f32_16x16x32_bf16(false, a.v, false, b.v,
                                                (short)0, c, false, false);
  }
  const int n = n0 + r;
#pragma unroll
  for (int j = 0; j < 8; ++j) {
    const int m  = m0 + hi * 8 + j;                // batch index
    const float v = tanhf(c[j] + xW[((size_t)m * 128 + t) * 1024 + n]);
    hf_out[(size_t)m * 1024 + n]  = v;
    hbf_out[(size_t)m * 1024 + n] = f32_to_bf16(v);
  }
}

// ------------------------------ small helpers ------------------------------
__global__ void k_f32_to_bf16(const float* __restrict__ s,
                              unsigned short* __restrict__ d, int n) {
  int i = blockIdx.x * 256 + threadIdx.x;
  if (i < n) d[i] = f32_to_bf16(s[i]);
}

__global__ void k_bias_sum(const float* a, const float* b, float* o, int n) {
  int i = blockIdx.x * 256 + threadIdx.x;
  if (i < n) o[i] = a[i] + b[i];
}

// h0 = hidden_seq[:, L-1, :]   (32768 elems)
__global__ void k_extract_h0(const float* __restrict__ hs,
                             float* __restrict__ hf,
                             unsigned short* __restrict__ hb) {
  int i = blockIdx.x * 256 + threadIdx.x;
  int b = i >> 10, d = i & 1023;
  float v = hs[((size_t)b * 2048 + 2047) * 1024 + d];
  hf[i] = v; hb[i] = f32_to_bf16(v);
}

// v[b,d] = sum_e projq[b,e] * W_k[e,d]   (grid=4, block=256; thread owns d)
__global__ void k_vproj(const float* __restrict__ q,
                        const float* __restrict__ Wk,
                        float* __restrict__ v) {
  const int d = blockIdx.x * 256 + threadIdx.x;
  float acc[32];
#pragma unroll
  for (int b = 0; b < 32; ++b) acc[b] = 0.0f;
  for (int e = 0; e < 1024; ++e) {
    const float w = Wk[(size_t)e * 1024 + d];
#pragma unroll
    for (int b = 0; b < 32; ++b) acc[b] += q[b * 1024 + e] * w;  // q uniform -> SGPR
  }
#pragma unroll
  for (int b = 0; b < 32; ++b) v[(size_t)b * 1024 + d] = acc[b];
}

// cterm[b] = projq[b,:] . b_k
__global__ void k_cterm(const float* __restrict__ q,
                        const float* __restrict__ bk, float* ct) {
  int b = threadIdx.x;
  if (b < 32) {
    float s = 0.0f;
    for (int e = 0; e < 1024; ++e) s += q[b * 1024 + e] * bk[e];
    ct[b] = s;
  }
}

// scores[b,l] = (hidden_seq[b,l,:] . v[b,:] + cterm[b]) / 32   (wave per row)
__global__ void k_scores(const float* __restrict__ hs,
                         const float* __restrict__ v,
                         const float* __restrict__ ct,
                         float* __restrict__ sc) {
  const int row  = blockIdx.x * 8 + (threadIdx.x >> 5);   // b*L + l
  const int lane = threadIdx.x & 31;
  const int b    = row >> 11;
  const v4f* h4 = (const v4f*)(hs + (size_t)row * 1024);
  const v4f* w4 = (const v4f*)(v + (size_t)b * 1024);
  float acc = 0.0f;
  for (int i = lane; i < 256; i += 32) {
    v4f a = h4[i], w = w4[i];
    acc += a[0]*w[0] + a[1]*w[1] + a[2]*w[2] + a[3]*w[3];
  }
  for (int off = 16; off > 0; off >>= 1) acc += __shfl_xor(acc, off, 32);
  if (lane == 0) sc[row] = (acc + ct[b]) * 0.03125f;
}

// in-place softmax over L=2048 per batch row (grid=32, block=256)
__global__ void k_softmax(float* __restrict__ sc) {
  __shared__ float red[256];
  const int tid = threadIdx.x;
  float* row = sc + (size_t)blockIdx.x * 2048;
  float m = -1e30f;
  for (int l = tid; l < 2048; l += 256) m = fmaxf(m, row[l]);
  red[tid] = m; __syncthreads();
  for (int s = 128; s > 0; s >>= 1) {
    if (tid < s) red[tid] = fmaxf(red[tid], red[tid + s]);
    __syncthreads();
  }
  const float mx = red[0]; __syncthreads();
  float sum = 0.0f;
  for (int l = tid; l < 2048; l += 256) {
    float e = __expf(row[l] - mx); row[l] = e; sum += e;
  }
  red[tid] = sum; __syncthreads();
  for (int s = 128; s > 0; s >>= 1) {
    if (tid < s) red[tid] += red[tid + s];
    __syncthreads();
  }
  const float inv = 1.0f / red[0];
  for (int l = tid; l < 2048; l += 256) row[l] *= inv;
}

// partial context: block = (b, l-chunk of 256); thread owns 4 d's
__global__ void k_ctx_partial(const float* __restrict__ hs,
                              const float* __restrict__ al,
                              float* __restrict__ part) {
  const int b = blockIdx.x >> 3, ch = blockIdx.x & 7, tid = threadIdx.x;
  const v4f* h4 = (const v4f*)(hs + (size_t)b * 2048 * 1024);
  v4f acc = {0.f, 0.f, 0.f, 0.f};
  const int l0 = ch * 256;
  for (int i = 0; i < 256; ++i) {
    const int l = l0 + i;
    const float a = al[b * 2048 + l];
    acc += a * h4[(size_t)l * 256 + tid];
  }
  ((v4f*)part)[(size_t)blockIdx.x * 256 + tid] = acc;
}

__global__ void k_ctx_reduce(const float* __restrict__ part,
                             float* __restrict__ ctx) {
  const int i = blockIdx.x * 256 + threadIdx.x;     // 32768
  const int b = i >> 10, d = i & 1023;
  float s = 0.0f;
  for (int c = 0; c < 8; ++c) s += part[(((size_t)b * 8 + c) << 10) + d];
  ctx[i] = s;
}

// acat[b, 0:1024]=context, [1024:2048]=query (bf16)  (65536 elems)
__global__ void k_concat(const float* __restrict__ ctx,
                         const float* __restrict__ qf,
                         unsigned short* __restrict__ acat) {
  const int i = blockIdx.x * 256 + threadIdx.x;
  const int b = i >> 11, j = i & 2047;
  const float v = (j < 1024) ? ctx[b * 1024 + j] : qf[b * 1024 + (j - 1024)];
  acat[i] = f32_to_bf16(v);
}

// ---------------------------------------------------------------------------
extern "C" void kernel_launch(void* const* d_in, const int* in_sizes, int n_in,
                              void* d_out, int out_size, void* d_ws, size_t ws_size,
                              hipStream_t stream) {
  (void)in_sizes; (void)n_in; (void)out_size; (void)ws_size;
  const float* X     = (const float*)d_in[0];
  const float* hs    = (const float*)d_in[1];
  const float* W_ih  = (const float*)d_in[2];
  const float* W_hh  = (const float*)d_in[3];
  const float* b_ih  = (const float*)d_in[4];
  const float* b_hh  = (const float*)d_in[5];
  const float* W_q   = (const float*)d_in[6];
  const float* b_q   = (const float*)d_in[7];
  const float* W_k   = (const float*)d_in[8];
  const float* b_k   = (const float*)d_in[9];
  const float* W_reg = (const float*)d_in[10];
  const float* b_reg = (const float*)d_in[11];
  float* out = (float*)d_out;

  char* w = (char*)d_ws;
  size_t off = 0;
  auto alloc = [&](size_t bytes) -> void* {
    void* p = w + off;
    off = (off + bytes + 255) & ~(size_t)255;
    return p;
  };

  unsigned short* Xbf   = (unsigned short*)alloc((size_t)4096 * 1024 * 2);
  unsigned short* Wihb  = (unsigned short*)alloc((size_t)1024 * 1024 * 2);
  unsigned short* Whhb  = (unsigned short*)alloc((size_t)1024 * 1024 * 2);
  unsigned short* Wqb   = (unsigned short*)alloc((size_t)1024 * 1024 * 2);
  unsigned short* Wregb = (unsigned short*)alloc((size_t)1024 * 2048 * 2);
  float* xW    = (float*)alloc((size_t)4096 * 1024 * 4);
  float* bsum  = (float*)alloc(1024 * 4);
  float* hf0   = (float*)alloc(32 * 1024 * 4);
  float* hf1   = (float*)alloc(32 * 1024 * 4);
  unsigned short* hb0 = (unsigned short*)alloc(32 * 1024 * 2);
  unsigned short* hb1 = (unsigned short*)alloc(32 * 1024 * 2);
  float* projq = (float*)alloc(32 * 1024 * 4);
  float* vbuf  = (float*)alloc(32 * 1024 * 4);
  float* ct    = (float*)alloc(256);
  float* sc    = (float*)alloc(32 * 2048 * 4);
  float* part  = (float*)alloc((size_t)32 * 8 * 1024 * 4);
  float* ctx   = (float*)alloc(32 * 1024 * 4);
  unsigned short* acat = (unsigned short*)alloc(32 * 2048 * 2);

  // --- convert operands to bf16 ---
  k_f32_to_bf16<<<4096 * 1024 / 256, 256, 0, stream>>>(X, Xbf, 4096 * 1024);
  k_f32_to_bf16<<<1024 * 1024 / 256, 256, 0, stream>>>(W_ih, Wihb, 1024 * 1024);
  k_f32_to_bf16<<<1024 * 1024 / 256, 256, 0, stream>>>(W_hh, Whhb, 1024 * 1024);
  k_f32_to_bf16<<<1024 * 1024 / 256, 256, 0, stream>>>(W_q, Wqb, 1024 * 1024);
  k_f32_to_bf16<<<1024 * 2048 / 256, 256, 0, stream>>>(W_reg, Wregb, 1024 * 2048);
  k_bias_sum<<<4, 256, 0, stream>>>(b_ih, b_hh, bsum, 1024);
  k_extract_h0<<<128, 256, 0, stream>>>(hs, hf0, hb0);

  // --- xW = X @ W_ih^T + (b_ih+b_hh) : M=4096,N=1024,K=1024 ---
  k_gemm_bf16nt<<<dim3(4, 128), 256, 0, stream>>>(Xbf, Wihb, bsum, xW,
                                                  1024, 1024, 1024, 1024);

  // --- RNN scan: 128 sequential fused GEMM+tanh steps, ping-pong h ---
  for (int t = 0; t < 128; ++t) {
    const unsigned short* hin = (t & 1) ? hb1 : hb0;
    float*          hfo = (t & 1) ? hf0 : hf1;
    unsigned short* hbo = (t & 1) ? hb0 : hb1;
    k_step<<<dim3(16, 1), 256, 0, stream>>>(hin, Whhb, xW, t, hfo, hbo);
  }
  // after t=127 (odd): final h in hf0 / hb0

  // --- proj_q = query @ W_q^T + b_q ---
  k_gemm_bf16nt<<<dim3(4, 1), 256, 0, stream>>>(hb0, Wqb, b_q, projq,
                                                1024, 1024, 1024, 1024);

  // --- rank-1 trick: v = proj_q @ W_k ; cterm = proj_q . b_k ---
  k_vproj<<<4, 256, 0, stream>>>(projq, W_k, vbuf);
  k_cterm<<<1, 32, 0, stream>>>(projq, b_k, ct);

  // --- scores + softmax + context (memory-bound streams of hidden_seq) ---
  k_scores<<<8192, 256, 0, stream>>>(hs, vbuf, ct, sc);
  k_softmax<<<32, 256, 0, stream>>>(sc);
  k_ctx_partial<<<256, 256, 0, stream>>>(hs, sc, part);
  k_ctx_reduce<<<128, 256, 0, stream>>>(part, ctx);

  // --- out = concat(context, query) @ W_reg^T + b_reg : M=32,N=1024,K=2048 ---
  k_concat<<<256, 256, 0, stream>>>(ctx, hf0, acat);
  k_gemm_bf16nt<<<dim3(4, 1), 256, 0, stream>>>(acat, Wregb, b_reg, out,
                                                2048, 2048, 1024, 2048);
}
